// Casual_self_Attention_5033701670934
// MI455X (gfx1250) — matrix operand order repaired and checked by reading, not compiled, not run on hardware
//
#include <hip/hip_runtime.h>

// ---------------------------------------------------------------------------
// Causal self-attention, bf16 WMMA (gfx1250 / MI455X)
//   x [4,4096,1024] f32, Wq/Wk/Wv [1024,1024] f32 -> out [4,4096,1024] f32
// ---------------------------------------------------------------------------

#define BATCH 4
#define SEQ   4096
#define DIM   1024
#define MTOT  (BATCH * SEQ)   // 16384 rows

typedef __bf16 bf16_t;
typedef __attribute__((ext_vector_type(16))) __bf16 v16bf;
typedef __attribute__((ext_vector_type(8)))  __bf16 v8bf;
typedef __attribute__((ext_vector_type(8)))  float  v8f;

union V16U { v16bf v; v8bf h[2]; };

// round-to-nearest-even f32 -> bf16 bit pattern
static __device__ __forceinline__ unsigned short f2bf(float f) {
  unsigned int u = __float_as_uint(f);
  u += 0x7FFFu + ((u >> 16) & 1u);
  return (unsigned short)(u >> 16);
}

// assemble a 16-element bf16 fragment from two contiguous 16B chunks
static __device__ __forceinline__ v16bf load_frag(const unsigned short* p0,
                                                  const unsigned short* p1) {
  V16U u;
  u.h[0] = *(const v8bf*)p0;
  u.h[1] = *(const v8bf*)p1;
  return u.v;
}

static __device__ __forceinline__ v8f wmma_bf16(v16bf a, v16bf b, v8f c) {
  return __builtin_amdgcn_wmma_f32_16x16x32_bf16(
      /*neg_a=*/false, a, /*neg_b=*/false, b,
      /*c_mod=*/(short)0, c, /*reuse_a=*/false, /*reuse_b=*/false);
}

// ---------------------------------------------------------------------------
// fp32 -> bf16 conversion (vectorized)
// ---------------------------------------------------------------------------
__global__ void cvt4(const float4* __restrict__ src,
                     ushort4* __restrict__ dst, int n4) {
  int i = blockIdx.x * blockDim.x + threadIdx.x;
  int stride = gridDim.x * blockDim.x;
  for (; i < n4; i += stride) {
    float4 f = src[i];
    dst[i] = make_ushort4(f2bf(f.x), f2bf(f.y), f2bf(f.z), f2bf(f.w));
  }
}

// ---------------------------------------------------------------------------
// QKV projection GEMM: C[m,n] = sum_d X[m,d] * W[n,d]   (X, W row-major bf16)
// One 32x64 C tile per wave (8 WMMAs / K-step); K-loop in steps of 32 on
// v_wmma_f32_16x16x32_bf16. Addresses computed inline from the kernarg base
// (keeps GLOBAL address-space inference -> global_load_b128, not flat), with
// unconditional speculative global_prefetch of the next K-tile.
// mat 0 -> Q (row-major), mat 1 -> K (row-major), mat 2 -> V transposed:
//   Vt[b][feature][seq]  so attention's P*V B-fragments are contiguous.
// ---------------------------------------------------------------------------
__global__ void __launch_bounds__(256)
qkv_proj(const unsigned short* __restrict__ xb,
         const unsigned short* __restrict__ wq,
         const unsigned short* __restrict__ wk,
         const unsigned short* __restrict__ wv,
         unsigned short* __restrict__ Qo,
         unsigned short* __restrict__ Ko,
         unsigned short* __restrict__ Vt) {
  const int lane = threadIdx.x & 31;
  const int lrow = lane & 15;
  const int lhi  = lane >> 4;

  const int wid  = blockIdx.x * (blockDim.x >> 5) + (threadIdx.x >> 5);
  const int tiles_per_mat = (MTOT / 32) * (DIM / 64);   // 8192
  const int mat = wid / tiles_per_mat;                  // 0..2
  const int rem = wid % tiles_per_mat;
  const int m0 = (rem / (DIM / 64)) * 32;
  const int n0 = (rem % (DIM / 64)) * 64;

  const unsigned short* W = (mat == 0) ? wq : (mat == 1) ? wk : wv;

  v8f acc[2][4] = {};

  for (int k0 = 0; k0 < DIM; k0 += 32) {
    v16bf a[2], b[4];
#pragma unroll
    for (int t = 0; t < 2; ++t) {
      // A-fragment (16x32): lane half -> K 0..15 / 16..31 (contiguous)
      const unsigned short* ap =
          xb + (size_t)(m0 + t * 16 + lrow) * DIM + k0 + lhi * 16;
      a[t] = load_frag(ap, ap + 8);
      __builtin_prefetch(ap + 32, 0, 1);   // speculative, branch-free
    }
#pragma unroll
    for (int t = 0; t < 4; ++t) {
      // B-fragment (32x16): lane half -> K 0..15 / 16..31 (contiguous)
      const unsigned short* bp =
          W + (size_t)(n0 + t * 16 + lrow) * DIM + k0 + lhi * 16;
      b[t] = load_frag(bp, bp + 8);
      __builtin_prefetch(bp + 32, 0, 1);
    }
#pragma unroll
    for (int i = 0; i < 2; ++i)
#pragma unroll
      for (int j = 0; j < 4; ++j)
        acc[i][j] = wmma_bf16(a[i], b[j], acc[i][j]);
  }

  // C layout: VGPR r -> row (r + 8*lanehalf), lane&15 -> col
  if (mat < 2) {
    unsigned short* O = (mat == 0) ? Qo : Ko;
#pragma unroll
    for (int i = 0; i < 2; ++i)
#pragma unroll
      for (int j = 0; j < 4; ++j)
#pragma unroll
        for (int r = 0; r < 8; ++r) {
          int row = m0 + i * 16 + r + 8 * lhi;
          int col = n0 + j * 16 + lrow;
          O[(size_t)row * DIM + col] = f2bf(acc[i][j][r]);
        }
  } else {
#pragma unroll
    for (int i = 0; i < 2; ++i)
#pragma unroll
      for (int j = 0; j < 4; ++j)
#pragma unroll
        for (int r = 0; r < 8; ++r) {
          int row = m0 + i * 16 + r + 8 * lhi;   // global m = b*SEQ + seq
          int col = n0 + j * 16 + lrow;          // feature
          int bb  = row >> 12;                   // / SEQ
          int seq = row & (SEQ - 1);
          Vt[(size_t)bb * DIM * SEQ + (size_t)col * SEQ + seq] =
              f2bf(acc[i][j][r]);
        }
  }
}

// ---------------------------------------------------------------------------
// Flash attention, causal, 32 queries per workgroup (8 waves).
//  - S = Q K^T: 1024-dim reduction split 128 dims/wave, reduced via ds_add_f32
//  - online softmax bookkeeping (m, l, alpha) in LDS by wave 0
//  - P*V: each wave owns a 128-feature output slice, f32 accumulators
//  - speculative global_prefetch of the next key block's K/V (K/V fit in L2)
// ---------------------------------------------------------------------------
__global__ void __launch_bounds__(256)
attn(const unsigned short* __restrict__ Q,
     const unsigned short* __restrict__ K,
     const unsigned short* __restrict__ Vt,
     float* __restrict__ out) {
  __shared__ float Sm[32][33];                       // score tile (f32)
  __shared__ float mvec[32], lvec[32], avec[32];     // online-softmax state
  __shared__ __align__(16) unsigned short Pm[32][40];// P tile (bf16), 16B rows

  const int tid  = threadIdx.x;
  const int lane = tid & 31;
  const int w    = tid >> 5;         // wave 0..7
  const int lrow = lane & 15;
  const int lhi  = lane >> 4;

  const int b  = blockIdx.x >> 7;    // / (SEQ/32)
  const int qb = blockIdx.x & 127;
  const int q0 = qb * 32;
  const int f0 = w * 128;            // this wave's feature/dim slice
  const float scale = 0.03125f;      // 1/sqrt(1024)

  const unsigned short* Qb = Q  + (size_t)b * SEQ * DIM;
  const unsigned short* Kb = K  + (size_t)b * SEQ * DIM;
  const unsigned short* Vb = Vt + (size_t)b * DIM * SEQ;

  // Q fragments for this wave's 128-dim slice: [m-tile][k-step], resident
  v16bf qf[2][4];
#pragma unroll
  for (int t = 0; t < 2; ++t)
#pragma unroll
    for (int ks = 0; ks < 4; ++ks) {
      const unsigned short* ap =
          Qb + (size_t)(q0 + t * 16 + lrow) * DIM + f0 + ks * 32;
      qf[t][ks] = load_frag(ap + lhi * 16, ap + lhi * 16 + 8);
    }

  v8f acc[2][8] = {};   // 32 queries x 128 features

  if (tid < 32) { mvec[tid] = -INFINITY; lvec[tid] = 0.0f; }
  __syncthreads();

  for (int kb = 0; kb <= qb; ++kb) {
    // speculative prefetch of next key block (harmless past the end)
    __builtin_prefetch(
        Kb + (size_t)((kb + 1) * 32 + lrow) * DIM + f0 + lhi * 64, 0, 1);
    __builtin_prefetch(
        Vb + (size_t)(f0 + lane * 4) * SEQ + (kb + 1) * 32, 0, 1);

    // zero score tile
    for (int idx = tid; idx < 32 * 33; idx += 256) (&Sm[0][0])[idx] = 0.0f;
    __syncthreads();

    // ---- partial S over this wave's 128 dims ----
    v8f sp[2][2] = {};
#pragma unroll
    for (int ks = 0; ks < 4; ++ks) {
      v16bf kf[2];
#pragma unroll
      for (int t = 0; t < 2; ++t) {
        const unsigned short* bp =
            Kb + (size_t)(kb * 32 + t * 16 + lrow) * DIM + f0 + ks * 32;
        kf[t] = load_frag(bp + lhi * 16, bp + lhi * 16 + 8);
      }
#pragma unroll
      for (int i = 0; i < 2; ++i)
#pragma unroll
        for (int j = 0; j < 2; ++j)
          sp[i][j] = wmma_bf16(qf[i][ks], kf[j], sp[i][j]);
    }
#pragma unroll
    for (int i = 0; i < 2; ++i)
#pragma unroll
      for (int j = 0; j < 2; ++j)
#pragma unroll
        for (int r = 0; r < 8; ++r)
          atomicAdd(&Sm[i * 16 + r + 8 * lhi][j * 16 + lrow], sp[i][j][r]);
    __syncthreads();

    // ---- online softmax (one row per lane of wave 0) ----
    if (tid < 32) {
      const int i  = tid;
      const int qg = q0 + i;
      float mo = mvec[i];
      float bm = -INFINITY;
      float sv[32];
#pragma unroll
      for (int k = 0; k < 32; ++k) {
        int kg = kb * 32 + k;
        float s = (kg <= qg) ? Sm[i][k] * scale : -INFINITY;
        sv[k] = s;
        bm = fmaxf(bm, s);
      }
      float mn    = fmaxf(mo, bm);
      float alpha = __expf(mo - mn);
      float sum   = 0.0f;
#pragma unroll
      for (int k = 0; k < 32; ++k) {
        float p = (sv[k] > -INFINITY) ? __expf(sv[k] - mn) : 0.0f;
        sum += p;
        Pm[i][k] = f2bf(p);
      }
      lvec[i] = lvec[i] * alpha + sum;
      mvec[i] = mn;
      avec[i] = alpha;
    }
    __syncthreads();

    // ---- rescale accumulators, then acc += P * V_slice ----
    v16bf pf[2];
#pragma unroll
    for (int t = 0; t < 2; ++t) {
      const unsigned short* pp = &Pm[t * 16 + lrow][0];
      pf[t] = load_frag(pp + lhi * 16, pp + lhi * 16 + 8);  // ds_load_b128 x2
    }
#pragma unroll
    for (int i = 0; i < 2; ++i)
#pragma unroll
      for (int r = 0; r < 8; ++r) {
        float al = avec[i * 16 + r + 8 * lhi];
#pragma unroll
        for (int j = 0; j < 8; ++j) acc[i][j][r] *= al;
      }
#pragma unroll
    for (int j = 0; j < 8; ++j) {
      const unsigned short* vp =
          Vb + (size_t)(f0 + j * 16 + lrow) * SEQ + kb * 32;
      v16bf vf = load_frag(vp + lhi * 16, vp + lhi * 16 + 8);
#pragma unroll
      for (int i = 0; i < 2; ++i)
        acc[i][j] = wmma_bf16(pf[i], vf, acc[i][j]);
    }
    __syncthreads();
  }

  // ---- epilogue: divide by softmax denominator, store f32 ----
#pragma unroll
  for (int i = 0; i < 2; ++i)
#pragma unroll
    for (int r = 0; r < 8; ++r) {
      int row = i * 16 + r + 8 * lhi;
      float inv = 1.0f / lvec[row];
#pragma unroll
      for (int j = 0; j < 8; ++j) {
        out[((size_t)b * SEQ + q0 + row) * DIM + f0 + j * 16 + lrow] =
            acc[i][j][r] * inv;
      }
    }
}

// ---------------------------------------------------------------------------
// launch
// ---------------------------------------------------------------------------
extern "C" void kernel_launch(void* const* d_in, const int* in_sizes, int n_in,
                              void* d_out, int out_size, void* d_ws,
                              size_t ws_size, hipStream_t stream) {
  (void)in_sizes; (void)n_in; (void)out_size; (void)ws_size;

  const float* x  = (const float*)d_in[0];
  const float* Wq = (const float*)d_in[1];
  const float* Wk = (const float*)d_in[2];
  const float* Wv = (const float*)d_in[3];

  unsigned short* ws  = (unsigned short*)d_ws;
  unsigned short* xb  = ws;                                  // 16384x1024
  unsigned short* wqb = xb  + (size_t)MTOT * DIM;            // 1024x1024
  unsigned short* wkb = wqb + (size_t)DIM * DIM;
  unsigned short* wvb = wkb + (size_t)DIM * DIM;
  unsigned short* Qb  = wvb + (size_t)DIM * DIM;             // 16384x1024
  unsigned short* Kb  = Qb  + (size_t)MTOT * DIM;
  unsigned short* Vt  = Kb  + (size_t)MTOT * DIM;            // [b][f][seq]

  cvt4<<<2048, 256, 0, stream>>>((const float4*)x,  (ushort4*)xb,
                                 MTOT * DIM / 4);
  cvt4<<<512,  256, 0, stream>>>((const float4*)Wq, (ushort4*)wqb,
                                 DIM * DIM / 4);
  cvt4<<<512,  256, 0, stream>>>((const float4*)Wk, (ushort4*)wkb,
                                 DIM * DIM / 4);
  cvt4<<<512,  256, 0, stream>>>((const float4*)Wv, (ushort4*)wvb,
                                 DIM * DIM / 4);

  // 3 mats * (16384/32)*(1024/64) tiles = 24576 waves / 8 waves-per-block
  qkv_proj<<<3072, 256, 0, stream>>>(xb, wqb, wkb, wvb, Qb, Kb, Vt);

  // BATCH * SEQ/32 query blocks
  attn<<<BATCH * (SEQ / 32), 256, 0, stream>>>(Qb, Kb, Vt, (float*)d_out);
}
